// Block_23862838296728
// MI455X (gfx1250) — matrix operand order repaired and checked by reading, not compiled, not run on hardware
//
#include <hip/hip_runtime.h>
#include <hip/hip_bf16.h>

typedef __attribute__((ext_vector_type(16))) _Float16 v16h;
typedef __attribute__((ext_vector_type(4)))  _Float16 v4h;
typedef __attribute__((ext_vector_type(8)))  float    v8f;
typedef unsigned int uint4v __attribute__((ext_vector_type(4)));
typedef int          int4v  __attribute__((ext_vector_type(4)));
typedef int          int8v  __attribute__((ext_vector_type(8)));

#define WMMA(a, b, c) __builtin_amdgcn_wmma_f32_16x16x32_f16(false, (a), false, (b), (short)0, (c), false, false)

#if __has_builtin(__builtin_amdgcn_tensor_load_to_lds) && __has_builtin(__builtin_amdgcn_s_wait_tensorcnt)
#define USE_TDM 1
#else
#define USE_TDM 0
#endif

// ---------------------------------------------------------------------------
// Build a v16h fragment from two 16-byte runs (forces b128 loads).
// ---------------------------------------------------------------------------
__device__ __forceinline__ v16h ld_frag_2x8(const _Float16* p0, const _Float16* p1) {
    float4 lo = *(const float4*)p0;
    float4 hi = *(const float4*)p1;
    const _Float16* a = (const _Float16*)&lo;
    const _Float16* b = (const _Float16*)&hi;
    v16h r;
#pragma unroll
    for (int j = 0; j < 8; ++j) { r[j] = a[j]; r[j + 8] = b[j]; }
    return r;
}

#if USE_TDM
// ---------------------------------------------------------------------------
// TDM 2-D tile load, global(f16)->LDS, with LDS padding of 4 DWORDs every
// 16 DWORDs (row of 32 halves -> stored stride 40 halves). ISA 8.3/8.4.
// ---------------------------------------------------------------------------
__device__ __forceinline__ void tdm_load_tile_f16(unsigned lds_byte,
                                                  const _Float16* gptr,
                                                  unsigned tile_d0, unsigned tile_d1,
                                                  unsigned tensor_d0, unsigned tensor_d1,
                                                  unsigned stride_elems) {
    unsigned long long ga = (unsigned long long)(size_t)(const void*)gptr;
    uint4v g0;
    g0[0] = 1u;                                            // count=1, user descriptor
    g0[1] = lds_byte;                                      // lds_addr
    g0[2] = (unsigned)(ga & 0xffffffffull);                // global_addr[31:0]
    g0[3] = (unsigned)((ga >> 32) & 0x1ffffffu)            // global_addr[56:32]
          | (2u << 30);                                    // type=2 ("image")
    int8v g1;
    g1[0] = (1 << 16)                                      // data_size = 2 bytes
          | (1 << 20)                                      // pad_enable
          | (3 << 22)                                      // pad_interval: 16 DWORDs
          | (3 << 25);                                     // pad_amount: 4 DWORDs
    g1[1] = (int)((tensor_d0 & 0xffffu) << 16);            // tensor_dim0[15:0]
    g1[2] = (int)((tensor_d0 >> 16) | ((tensor_d1 & 0xffffu) << 16));
    g1[3] = (int)((tensor_d1 >> 16) | (tile_d0 << 16));    // tile_dim0
    g1[4] = (int)tile_d1;                                  // tile_dim1 (tile_dim2=0)
    g1[5] = (int)stride_elems;                             // tensor_dim0_stride[31:0]
    g1[6] = 0;
    g1[7] = 0;
    int4v z4 = {0, 0, 0, 0};
#if defined(__clang_major__) && (__clang_major__ >= 23)
    int8v z8 = {0, 0, 0, 0, 0, 0, 0, 0};
    __builtin_amdgcn_tensor_load_to_lds(g0, g1, z4, z4, z8, 0);
#else
    __builtin_amdgcn_tensor_load_to_lds(g0, g1, z4, z4, 0);
#endif
}
#endif

// ---------------------------------------------------------------------------
// LayerNorm: one block per row of 1024, fp32 in -> f16 out (vectorized)
// ---------------------------------------------------------------------------
__global__ __launch_bounds__(256) void ln_kernel(const float* __restrict__ x,
                                                 const float* __restrict__ w,
                                                 const float* __restrict__ b,
                                                 _Float16* __restrict__ out) {
    const int D = 1024;
    int row = blockIdx.x;
    int t = threadIdx.x, lane = t & 31, wave = t >> 5;
    float4 v = ((const float4*)(x + (size_t)row * D))[t];
    float s = v.x + v.y + v.z + v.w;
    float s2 = v.x * v.x + v.y * v.y + v.z * v.z + v.w * v.w;
#pragma unroll
    for (int off = 16; off > 0; off >>= 1) {
        s  += __shfl_xor(s,  off, 32);
        s2 += __shfl_xor(s2, off, 32);
    }
    __shared__ float rs[8], rq[8];
    if (lane == 0) { rs[wave] = s; rq[wave] = s2; }
    __syncthreads();
    float ts = 0.f, tq = 0.f;
#pragma unroll
    for (int i = 0; i < 8; ++i) { ts += rs[i]; tq += rq[i]; }
    float mu = ts * (1.0f / D);
    float var = tq * (1.0f / D) - mu * mu;
    float inv = rsqrtf(var + 1e-5f);
    float4 wg = ((const float4*)w)[t];
    float4 bg = ((const float4*)b)[t];
    v4h o;
    o[0] = (_Float16)((v.x - mu) * inv * wg.x + bg.x);
    o[1] = (_Float16)((v.y - mu) * inv * wg.y + bg.y);
    o[2] = (_Float16)((v.z - mu) * inv * wg.z + bg.z);
    o[3] = (_Float16)((v.w - mu) * inv * wg.w + bg.w);
    *((v4h*)(out + (size_t)row * D) + t) = o;
}

// ---------------------------------------------------------------------------
// Tiled transpose + fp32->f16: in [K][N] -> out [N][K]
// ---------------------------------------------------------------------------
__global__ __launch_bounds__(256) void transpose_cvt(const float* __restrict__ in,
                                                     _Float16* __restrict__ out,
                                                     int K, int N) {
    __shared__ float tile[32][33];
    int tx = threadIdx.x, ty = threadIdx.y;
    int n = blockIdx.x * 32 + tx;
    int k0 = blockIdx.y * 32;
#pragma unroll
    for (int i = ty; i < 32; i += 8)
        tile[i][tx] = in[(size_t)(k0 + i) * N + n];
    __syncthreads();
    int k = k0 + tx;
    int n0 = blockIdx.x * 32;
#pragma unroll
    for (int i = ty; i < 32; i += 8)
        out[(size_t)(n0 + i) * K + k] = (_Float16)tile[tx][i];
}

// Wq/Wk/Wv [H,1024,64] fp32 -> fused transposed [3072][1024] f16
__global__ __launch_bounds__(256) void pack_qkv_w_t(const float* __restrict__ Wq,
                                                    const float* __restrict__ Wk,
                                                    const float* __restrict__ Wv,
                                                    _Float16* __restrict__ out) {
    __shared__ float tile[32][33];
    int tx = threadIdx.x, ty = threadIdx.y;
    int sec = blockIdx.z >> 4, h = blockIdx.z & 15;
    const float* W = (sec == 0) ? Wq : (sec == 1) ? Wk : Wv;
    const float* Wh = W + (size_t)h * 1024 * 64;   // [1024][64]
    int e = blockIdx.x * 32 + tx;
    int d0 = blockIdx.y * 32;
#pragma unroll
    for (int i = ty; i < 32; i += 8)
        tile[i][tx] = Wh[(size_t)(d0 + i) * 64 + e];
    __syncthreads();
    int d = d0 + tx;
    int e0 = blockIdx.x * 32;
#pragma unroll
    for (int i = ty; i < 32; i += 8)
        out[(size_t)(sec * 1024 + h * 64 + e0 + i) * 1024 + d] = (_Float16)tile[tx][i];
}

__global__ void pack_qkv_b(const float* __restrict__ bq, const float* __restrict__ bk,
                           const float* __restrict__ bv, float* __restrict__ out) {
    int col = blockIdx.x * 256 + threadIdx.x;
    if (col >= 3072) return;
    int sec = col >> 10, he = col & 1023;
    const float* B = (sec == 0) ? bq : (sec == 1) ? bk : bv;
    out[col] = B[he];
}

// ---------------------------------------------------------------------------
// WMMA GEMM: C[M,N] = A[M,K](f16) * Bt[N,K](f16, n-major) + bias (+res)(+gelu)
// block tile 64x256, 8 waves (2x4), each wave 32x64 = 2x4 accumulators.
// Tile staging: TDM DMA (zero VGPR traffic, HW-padded LDS rows) when
// available; else short-lived load+ds_store. LDS row stride = 40 halves.
// ---------------------------------------------------------------------------
__global__ __launch_bounds__(256) void gemm_kernel(const _Float16* __restrict__ A,
                                                   const _Float16* __restrict__ Bt,
                                                   const float* __restrict__ bias,
                                                   const float* __restrict__ residual,
                                                   float* __restrict__ outF,
                                                   _Float16* __restrict__ outH,
                                                   int M, int N, int K, int act) {
    __shared__ __align__(16) _Float16 sA[64 * 40];
    __shared__ __align__(16) _Float16 sB[256 * 40];
    int t = threadIdx.x, lane = t & 31, wave = t >> 5;
    int wm = wave & 1, wn = wave >> 1;               // 2 x 4 wave grid
    int g = lane >> 4, nlo = lane & 15;
    int m0 = blockIdx.y * 64, n0 = blockIdx.x * 256;

#if USE_TDM
    unsigned ldsA = (unsigned)(size_t)(void*)sA;
    unsigned ldsB = (unsigned)(size_t)(void*)sB;
    const _Float16* gAblk = A + (size_t)m0 * K;
    const _Float16* gBblk = Bt + (size_t)n0 * K;
#else
    int ar = t >> 2, ac = (t & 3) * 8;               // 4 chunks of 8 halves per row
    const _Float16* gA = A + (size_t)(m0 + ar) * K + ac;
    const _Float16* gB = Bt + (size_t)(n0 + ar) * K + ac;
    size_t bRowStep = (size_t)64 * K;                // 64 rows per staging pass
#endif

    v8f acc[2][4] = {};

    for (int k0 = 0; k0 < K; k0 += 32) {
#if USE_TDM
        if (wave == 0) {
            // A tile: 64 rows x 32 k ; B tile: 256 rows x 32 k (both padded to 40)
            tdm_load_tile_f16(ldsA, gAblk + k0, 32u, 64u,  (unsigned)K, (unsigned)M, (unsigned)K);
            tdm_load_tile_f16(ldsB, gBblk + k0, 32u, 256u, (unsigned)K, (unsigned)N, (unsigned)K);
            __builtin_amdgcn_s_wait_tensorcnt(0);
        }
        __syncthreads();
#else
        {
            float4 va = *(const float4*)(gA + k0);
            *(float4*)(sA + ar * 40 + ac) = va;
#pragma unroll
            for (int q = 0; q < 4; ++q) {
                float4 vb = *(const float4*)(gB + q * bRowStep + k0);
                *(float4*)(sB + (ar + 64 * q) * 40 + ac) = vb;
            }
        }
        __syncthreads();
#endif
        // fragments (all b128 LDS loads)
        v16h af[2], bf[4];
#pragma unroll
        for (int i = 0; i < 2; ++i) {
            const _Float16* r = sA + (wm * 32 + i * 16 + nlo) * 40;
            af[i] = ld_frag_2x8(r + g * 8, r + 16 + g * 8);
        }
#pragma unroll
        for (int j = 0; j < 4; ++j) {
            const _Float16* r = sB + (wn * 64 + j * 16 + nlo) * 40 + g * 8;
            bf[j] = ld_frag_2x8(r, r + 16);
        }
#pragma unroll
        for (int i = 0; i < 2; ++i)
#pragma unroll
            for (int j = 0; j < 4; ++j)
                acc[i][j] = WMMA(af[i], bf[j], acc[i][j]);
        __syncthreads();
    }

#pragma unroll
    for (int i = 0; i < 2; ++i) {
#pragma unroll
        for (int j = 0; j < 4; ++j) {
#pragma unroll
            for (int r = 0; r < 8; ++r) {
                int row = m0 + wm * 32 + i * 16 + r + g * 8;
                int col = n0 + wn * 64 + j * 16 + nlo;
                size_t idx = (size_t)row * N + col;
                float v = acc[i][j][r];
                if (bias)     v += bias[col];
                if (residual) v += residual[idx];
                if (act)      v = 0.5f * v * (1.0f + erff(v * 0.70710678118654752f));
                if (outF) outF[idx] = v;
                if (outH) outH[idx] = (_Float16)v;
            }
        }
    }
}

// ---------------------------------------------------------------------------
// Flash attention: 4 waves = 64 q rows of one (b,h); K tile [32key][72] and
// V tile staged TRANSPOSED [64hs][40key] so every fragment gather is b128.
// QKV layout: [B*S, 3072] f16, col = sec*1024 + h*64 + e. Out [B*S,1024] f16.
// ---------------------------------------------------------------------------
__global__ __launch_bounds__(128) void attn_kernel(const _Float16* __restrict__ QKV,
                                                   _Float16* __restrict__ out) {
    const int S = 2048, LD = 3072, nQB = S / 64;
    __shared__ __align__(16) _Float16 sK[32 * 72];
    __shared__ __align__(16) _Float16 sVt[64 * 40];
    __shared__ __align__(16) _Float16 sP[4][16 * 40];

    int bid = blockIdx.x;
    int qb = bid % nQB;
    int bh = bid / nQB;
    int h = bh % 16, b = bh / 16;
    int t = threadIdx.x, lane = t & 31, wave = t >> 5;
    int g = lane >> 4, nlo = lane & 15;
    int q0 = qb * 64 + wave * 16;
    size_t rowbase = (size_t)b * S;

    v16h aq0, aq1;
    {
        const _Float16* Qp = QKV + (rowbase + q0 + nlo) * LD + h * 64;
        aq0 = ld_frag_2x8(Qp + g * 8, Qp + 16 + g * 8);
        aq1 = ld_frag_2x8(Qp + 32 + g * 8, Qp + 48 + g * 8);
#pragma unroll
        for (int j = 0; j < 16; ++j) {
            aq0[j] = aq0[j] * (_Float16)0.125f;
            aq1[j] = aq1[j] * (_Float16)0.125f;
        }
    }

    v8f o0 = {}, o1 = {}, o2 = {}, o3 = {};
    float m8[8], l8[8];
#pragma unroll
    for (int r = 0; r < 8; ++r) { m8[r] = -1e30f; l8[r] = 0.f; }

    int nT = qb * 2 + 2;
    for (int kt = 0; kt < nT; ++kt) {
        int kb = kt * 32;
        {   // K tile: 32 keys x 64 hs, padded stride 72
            int rr = t >> 2, cc = (t & 3) * 16;
            const float4* gk = (const float4*)(QKV + (rowbase + kb + rr) * LD + 1024 + h * 64 + cc);
            *(float4*)(sK + rr * 72 + cc)     = gk[0];
            *(float4*)(sK + rr * 72 + cc + 8) = gk[1];
            // V tile transposed: sVt[hs][key], stride 40
#pragma unroll
            for (int q = 0; q < 2; ++q) {
                int c = t + 128 * q;
                int key = c >> 3, hc = (c & 7) * 8;
                float4 gv = *(const float4*)(QKV + (rowbase + kb + key) * LD + 2048 + h * 64 + hc);
                const _Float16* vv = (const _Float16*)&gv;
#pragma unroll
                for (int i = 0; i < 8; ++i)
                    sVt[(hc + i) * 40 + key] = vv[i];
            }
        }
        __syncthreads();

        if (kb <= q0 + 15) {
            v8f s0 = {}, s1 = {};
            {
                const _Float16* r0 = sK + nlo * 72 + g * 8;
                const _Float16* r1 = sK + (16 + nlo) * 72 + g * 8;
                s0 = WMMA(aq0, ld_frag_2x8(r0, r0 + 16), s0);
                s0 = WMMA(aq1, ld_frag_2x8(r0 + 32, r0 + 48), s0);
                s1 = WMMA(aq0, ld_frag_2x8(r1, r1 + 16), s1);
                s1 = WMMA(aq1, ld_frag_2x8(r1 + 32, r1 + 48), s1);
            }
            _Float16* P = &sP[wave][0];
#pragma unroll
            for (int r = 0; r < 8; ++r) {
                int m = q0 + r + g * 8;
                float v0 = s0[r]; if (kb + nlo > m)      v0 = -1e30f;
                float v1 = s1[r]; if (kb + 16 + nlo > m) v1 = -1e30f;
                float tmax = fmaxf(v0, v1);
#pragma unroll
                for (int off = 8; off > 0; off >>= 1)
                    tmax = fmaxf(tmax, __shfl_xor(tmax, off, 32));
                float mnew = fmaxf(m8[r], tmax);
                float p0 = __expf(v0 - mnew);
                float p1 = __expf(v1 - mnew);
                float rsum = p0 + p1;
#pragma unroll
                for (int off = 8; off > 0; off >>= 1)
                    rsum += __shfl_xor(rsum, off, 32);
                float corr = __expf(m8[r] - mnew);
                l8[r] = l8[r] * corr + rsum;
                m8[r] = mnew;
                o0[r] *= corr; o1[r] *= corr; o2[r] *= corr; o3[r] *= corr;
                int mm = r + g * 8;
                P[mm * 40 + nlo]      = (_Float16)p0;
                P[mm * 40 + 16 + nlo] = (_Float16)p1;
            }
            const _Float16* pr = P + nlo * 40;
            v16h ap = ld_frag_2x8(pr + g * 8, pr + 16 + g * 8);
            const _Float16* v0p = sVt + nlo * 40 + g * 8;
            o0 = WMMA(ap, ld_frag_2x8(v0p,           v0p + 16),           o0);
            o1 = WMMA(ap, ld_frag_2x8(v0p + 16 * 40, v0p + 16 * 40 + 16), o1);
            o2 = WMMA(ap, ld_frag_2x8(v0p + 32 * 40, v0p + 32 * 40 + 16), o2);
            o3 = WMMA(ap, ld_frag_2x8(v0p + 48 * 40, v0p + 48 * 40 + 16), o3);
        }
        __syncthreads();
    }

#pragma unroll
    for (int r = 0; r < 8; ++r) {
        float inv = 1.0f / l8[r];
        size_t row = rowbase + q0 + r + g * 8;
        _Float16* op = out + row * 1024 + h * 64;
        op[nlo]      = (_Float16)(o0[r] * inv);
        op[16 + nlo] = (_Float16)(o1[r] * inv);
        op[32 + nlo] = (_Float16)(o2[r] * inv);
        op[48 + nlo] = (_Float16)(o3[r] * inv);
    }
}

// ---------------------------------------------------------------------------
// Host orchestration
// ---------------------------------------------------------------------------
extern "C" void kernel_launch(void* const* d_in, const int* in_sizes, int n_in,
                              void* d_out, int out_size, void* d_ws, size_t ws_size,
                              hipStream_t stream) {
    (void)in_sizes; (void)n_in; (void)out_size; (void)ws_size;
    const int B = 2, S = 2048, D = 1024, Hh = 16, F = 4096;
    const int M = B * S;
    const int N3 = 3 * D;

    const float* x     = (const float*)d_in[0];
    const float* ln1_w = (const float*)d_in[1];
    const float* ln1_b = (const float*)d_in[2];
    const float* Wq    = (const float*)d_in[3];
    const float* bq    = (const float*)d_in[4];
    const float* Wk    = (const float*)d_in[5];
    const float* bk    = (const float*)d_in[6];
    const float* Wv    = (const float*)d_in[7];
    const float* bv    = (const float*)d_in[8];
    const float* Wp    = (const float*)d_in[9];
    const float* bp    = (const float*)d_in[10];
    const float* ln2_w = (const float*)d_in[11];
    const float* ln2_b = (const float*)d_in[12];
    const float* W1    = (const float*)d_in[13];
    const float* b1    = (const float*)d_in[14];
    const float* W2    = (const float*)d_in[15];
    const float* b2    = (const float*)d_in[16];
    float* out = (float*)d_out;

    char* p = (char*)d_ws;
    _Float16* hLN    = (_Float16*)p; p += (size_t)M * D * 2;
    _Float16* Wqkv_t = (_Float16*)p; p += (size_t)N3 * D * 2;
    float*    bqkv   = (float*)p;    p += (size_t)N3 * 4;
    _Float16* QKV    = (_Float16*)p; p += (size_t)M * N3 * 2;
    _Float16* attn_h = (_Float16*)p; p += (size_t)M * D * 2;
    _Float16* Wp_t   = (_Float16*)p; p += (size_t)D * D * 2;
    float*    x2     = (float*)p;    p += (size_t)M * D * 4;
    _Float16* h2     = (_Float16*)p; p += (size_t)M * D * 2;
    _Float16* W1_t   = (_Float16*)p; p += (size_t)F * D * 2;
    _Float16* ffn1_h = (_Float16*)p; p += (size_t)M * F * 2;
    _Float16* W2_t   = (_Float16*)p; p += (size_t)D * F * 2;

    dim3 tb(32, 8);
    ln_kernel<<<M, 256, 0, stream>>>(x, ln1_w, ln1_b, hLN);
    pack_qkv_w_t<<<dim3(2, 32, 48), tb, 0, stream>>>(Wq, Wk, Wv, Wqkv_t);
    pack_qkv_b<<<(N3 + 255) / 256, 256, 0, stream>>>(bq, bk, bv, bqkv);
    transpose_cvt<<<dim3(D / 32, D / 32), tb, 0, stream>>>(Wp, Wp_t, D, D);
    transpose_cvt<<<dim3(F / 32, D / 32), tb, 0, stream>>>(W1, W1_t, D, F);
    transpose_cvt<<<dim3(D / 32, F / 32), tb, 0, stream>>>(W2, W2_t, F, D);
    gemm_kernel<<<dim3(N3 / 256, M / 64), 256, 0, stream>>>(
        hLN, Wqkv_t, bqkv, nullptr, nullptr, QKV, M, N3, D, 0);
    attn_kernel<<<B * Hh * (S / 64), 128, 0, stream>>>(QKV, attn_h);
    gemm_kernel<<<dim3(D / 256, M / 64), 256, 0, stream>>>(
        attn_h, Wp_t, bp, x, x2, nullptr, M, D, D, 0);
    ln_kernel<<<M, 256, 0, stream>>>(x2, ln2_w, ln2_b, h2);
    gemm_kernel<<<dim3(F / 256, M / 64), 256, 0, stream>>>(
        h2, W1_t, b1, nullptr, nullptr, ffn1_h, M, F, D, 1);
    gemm_kernel<<<dim3(D / 256, M / 64), 256, 0, stream>>>(
        ffn1_h, W2_t, b2, x2, out, nullptr, M, D, F, 0);
}
